// RealStateSpaceModel_798863917309
// MI455X (gfx1250) — matrix operand order repaired and checked by reading, not compiled, not run on hardware
//
#include <hip/hip_runtime.h>

// ---------------------------------------------------------------------------
// MI455X (gfx1250) real-rotation SSM.
//
// Roofline: ~69 GFLOP of GEMM vs ~400 MB of HBM traffic (17 us @ 23.3 TB/s)
// => matrix-core bound. Both GEMMs run as bf16 v_wmma_f32_16x16x32_bf16 with
// f32 accumulation. Weights are pre-transposed (B->B^T, C->C^T) once so both
// GEMM operands are "contiguous along K" and can be staged into LDS with
// gfx1250 global_load_async_to_lds_b128 (ASYNCcnt) into double-buffered LDS:
// tile k+1 streams in while tile k feeds the WMMAs; one barrier per k-step.
// The 4096-step linear recurrence is trivial FLOP-wise -> f32 scan, 2048 lanes.
// ---------------------------------------------------------------------------

typedef __attribute__((ext_vector_type(16))) __bf16 v16bf;
typedef __attribute__((ext_vector_type(8)))  __bf16 v8bf;
typedef __attribute__((ext_vector_type(8)))  float  v8f;

__device__ __forceinline__ __bf16 f2bf(float f) {
  union { float f; unsigned u; } v; v.f = f;
  unsigned r = v.u + 0x7FFFu + ((v.u >> 16) & 1u);  // round-to-nearest-even
  unsigned short h = (unsigned short)(r >> 16);
  __bf16 out; __builtin_memcpy(&out, &h, 2);
  return out;
}

// gfx1250 async copy: per-lane 16B global -> LDS, tracked by ASYNCcnt.
__device__ __forceinline__ void async_ld16(unsigned lds_addr, const void* g) {
  unsigned long long ga = (unsigned long long)(size_t)g;
  asm volatile("global_load_async_to_lds_b128 %0, %1, off"
               :: "v"(lds_addr), "v"(ga) : "memory");
}
__device__ __forceinline__ void wait_async0() {
  asm volatile("s_wait_asynccnt 0" ::: "memory");
}
__device__ __forceinline__ unsigned lds_off(const void* p) {
  return (unsigned)(size_t)p;  // low 32 bits of a flat LDS pointer = DS offset
}

// ---------------------------------------------------------------------------
// Convert f32 [K][N] -> bf16 transposed [N][K] (for the B and C weights).
// ---------------------------------------------------------------------------
__global__ __launch_bounds__(256) void cvt_tr_kernel(const float* __restrict__ in,
                                                     __bf16* __restrict__ out,
                                                     int K, int N) {
  int i = blockIdx.x * 256 + threadIdx.x;
  if (i < K * N) {
    int k = i / N, n = i % N;
    out[(long)n * K + k] = f2bf(in[i]);
  }
}

// ---------------------------------------------------------------------------
// RMSNorm: one 256-thread block per row of 1024; writes bf16.
// ---------------------------------------------------------------------------
__global__ __launch_bounds__(256) void rmsnorm_kernel(const float* __restrict__ u,
                                                      const float* __restrict__ w,
                                                      __bf16* __restrict__ out, int D) {
  const long row = blockIdx.x;
  const float* up = u + row * (long)D;
  float vals[4];
  float s = 0.f;
#pragma unroll
  for (int i = 0; i < 4; ++i) {
    vals[i] = up[threadIdx.x + i * 256];
    s += vals[i] * vals[i];
  }
#pragma unroll
  for (int off = 16; off > 0; off >>= 1) s += __shfl_xor(s, off, 32);
  __shared__ float red[8];
  if ((threadIdx.x & 31) == 0) red[threadIdx.x >> 5] = s;
  __syncthreads();
  float tot = 0.f;
#pragma unroll
  for (int j = 0; j < 8; ++j) tot += red[j];
  const float scale = rsqrtf(tot / (float)D + 1e-6f);
#pragma unroll
  for (int i = 0; i < 4; ++i) {
    int c = threadIdx.x + i * 256;
    out[row * (long)D + c] = f2bf(vals[i] * scale * w[c]);
  }
}

// ---------------------------------------------------------------------------
// bf16 GEMM:  Cg[M,N](f32,rm) = A[M,K](bf16,rm) @ B[K,N], B given as BT[N][K].
// Block tile 128x128, 8 waves (wave32) as 4(M) x 2(N), each wave a 32x64
// register tile = 2x4 accumulators -> 8 x wmma 16x16x32 per k-step.
//
// Both LDS tiles are 128 rows x 32 K bf16 (row stride 40 to spread banks),
// double-buffered, filled by global_load_async_to_lds_b128 (2 chunks/thread
// per tile). Fragment gathers are contiguous ds_load_b128 matching the
// 16-bit WMMA operand layouts (CDNA5 ISA 7.12.2):
//   A frag, lane l (half=l>>4, r=l&15): row r, K half*8..+7 and 16+half*8..+7
//   B frag, lane l: column r, K half*8..+7 and 16+half*8..+7
// ---------------------------------------------------------------------------
__global__ __launch_bounds__(256) void gemm_bf16_kernel(const __bf16* __restrict__ A,
                                                        const __bf16* __restrict__ BT,
                                                        float* __restrict__ Cg,
                                                        int M, int N, int K) {
  __shared__ __attribute__((aligned(16))) __bf16 lsA[2][128 * 40];
  __shared__ __attribute__((aligned(16))) __bf16 lsB[2][128 * 40];

  const int tid   = threadIdx.x;
  const int lane  = tid & 31;
  const int wave  = tid >> 5;
  const int waveM = wave & 3;   // 0..3
  const int waveN = wave >> 2;  // 0..1
  const int half  = lane >> 4;  // 0/1
  const int lr    = lane & 15;

  const long mBase = (long)blockIdx.x * 128;
  const long nBase = (long)blockIdx.y * 128;

  // each thread owns chunks c = tid*2, tid*2+1 of the 512 x 16B chunks/tile
  const int c0row = (tid * 2) >> 2;          // 0..127
  const int c0kg  = (tid * 2) & 3;           // 0..3 (+1 chunk is c0kg+1)

  v8f acc[2][4] = {};

  auto stage = [&](int buf, int k0) {
#pragma unroll
    for (int i = 0; i < 2; ++i) {
      const int kg = c0kg + i;               // chunks are adjacent in kg
      async_ld16(lds_off(&lsA[buf][c0row * 40 + kg * 8]),
                 A + (mBase + c0row) * (long)K + k0 + kg * 8);
      async_ld16(lds_off(&lsB[buf][c0row * 40 + kg * 8]),
                 BT + (nBase + c0row) * (long)K + k0 + kg * 8);
    }
  };

  // prologue: tile 0 into buffer 0
  stage(0, 0);
  wait_async0();
  __syncthreads();

  int buf = 0;
  for (int k0 = 0; k0 < K; k0 += 32) {
    // stream next tile into the other buffer while computing this one
    if (k0 + 32 < K) stage(buf ^ 1, k0 + 32);

    // ---- gather WMMA fragments from LDS (current buffer)
    v16bf afrag[2];
#pragma unroll
    for (int mt = 0; mt < 2; ++mt) {
      const int row = waveM * 32 + mt * 16 + lr;
      const v8bf lo = *(const v8bf*)(&lsA[buf][row * 40 + half * 8]);
      const v8bf hi = *(const v8bf*)(&lsA[buf][row * 40 + 16 + half * 8]);
      afrag[mt] = __builtin_shufflevector(lo, hi, 0, 1, 2, 3, 4, 5, 6, 7,
                                          8, 9, 10, 11, 12, 13, 14, 15);
    }
    v16bf bfrag[4];
#pragma unroll
    for (int nt = 0; nt < 4; ++nt) {
      const int col = waveN * 64 + nt * 16 + lr;
      const v8bf lo = *(const v8bf*)(&lsB[buf][col * 40 + half * 8]);
      const v8bf hi = *(const v8bf*)(&lsB[buf][col * 40 + 16 + half * 8]);
      bfrag[nt] = __builtin_shufflevector(lo, hi, 0, 1, 2, 3, 4, 5, 6, 7,
                                          8, 9, 10, 11, 12, 13, 14, 15);
    }

    // ---- 8 x v_wmma_f32_16x16x32_bf16
#pragma unroll
    for (int mt = 0; mt < 2; ++mt)
#pragma unroll
      for (int nt = 0; nt < 4; ++nt)
        acc[mt][nt] = __builtin_amdgcn_wmma_f32_16x16x32_bf16(
            false, afrag[mt], false, bfrag[nt], (short)0, acc[mt][nt],
            false, false);

    wait_async0();     // next tile fully in LDS (this wave's copies)
    __syncthreads();   // everyone's copies visible; old buffer reusable
    buf ^= 1;
  }

  // ---- store C/D: VGPR i, lanes 0-15 -> M=i, lanes 16-31 -> M=8+i (7.12.2)
#pragma unroll
  for (int mt = 0; mt < 2; ++mt) {
#pragma unroll
    for (int nt = 0; nt < 4; ++nt) {
      const long rowb = mBase + waveM * 32 + mt * 16 + half * 8;
      const long col  = nBase + waveN * 64 + nt * 16 + lr;
#pragma unroll
      for (int i = 0; i < 8; ++i)
        Cg[(rowb + i) * (long)N + col] = acc[mt][nt][i];
    }
  }
}

// ---------------------------------------------------------------------------
// Sequential rotation scan:  x[t] = x[t-1] @ A_k + uB[t],  x[-1] = x0.
// grid = batch(8), block = 256 (one lane per k). Writes x as bf16 for GEMM2
// and the final state (f32) into d_out's tail.
// ---------------------------------------------------------------------------
__global__ __launch_bounds__(256) void scan_kernel(const float* __restrict__ uB,
                                                   const float* __restrict__ A,
                                                   const float* __restrict__ x0,
                                                   __bf16* __restrict__ xbf,
                                                   float* __restrict__ new_state,
                                                   int L, int K) {
  const int b = blockIdx.x;
  const int k = threadIdx.x;  // K == 256 == blockDim.x
  const float a00 = A[k * 4 + 0];
  const float a01 = A[k * 4 + 1];
  const float a10 = A[k * 4 + 2];
  const float a11 = A[k * 4 + 3];
  float xr = x0[((long)b * K + k) * 2 + 0];
  float xi = x0[((long)b * K + k) * 2 + 1];
  const float* up = uB + (long)b * L * K * 2 + k * 2;
  __bf16*      xp = xbf + (long)b * L * K * 2 + k * 2;
  const int stride = K * 2;
  for (int t = 0; t < L; ++t) {
    const float u0 = up[0];
    const float u1 = up[1];
    // row-vector update: x_j = sum_i x_i * A[i][j]  (einsum 'i,ij->j')
    const float nr = fmaf(xr, a00, fmaf(xi, a10, u0));
    const float ni = fmaf(xr, a01, fmaf(xi, a11, u1));
    xr = nr; xi = ni;
    xp[0] = f2bf(xr);
    xp[1] = f2bf(xi);
    up += stride;
    xp += stride;
  }
  new_state[((long)b * K + k) * 2 + 0] = xr;
  new_state[((long)b * K + k) * 2 + 1] = xi;
}

// ---------------------------------------------------------------------------
// Launch
// ---------------------------------------------------------------------------
extern "C" void kernel_launch(void* const* d_in, const int* in_sizes, int n_in,
                              void* d_out, int out_size, void* d_ws, size_t ws_size,
                              hipStream_t stream) {
  (void)in_sizes; (void)n_in; (void)out_size; (void)ws_size;

  constexpr int BATCH = 8, L = 4096, DIN = 1024, STATE = 512, DOUT = 1024;
  constexpr int K2 = STATE / 2;          // 256 rotation blocks
  constexpr long ROWS = (long)BATCH * L; // 32768

  const float* u  = (const float*)d_in[0];  // (8,4096,1024)
  const float* x0 = (const float*)d_in[1];  // (8,256,2)
  const float* A  = (const float*)d_in[2];  // (256,2,2)
  const float* B  = (const float*)d_in[3];  // (1024,512)
  const float* C  = (const float*)d_in[4];  // (512,1024)
  const float* nw = (const float*)d_in[5];  // (1024,)

  float* y         = (float*)d_out;                // (8,4096,1024)
  float* new_state = (float*)d_out + ROWS * DOUT;  // (8,256,2)

  // workspace layout
  char* ws = (char*)d_ws;
  __bf16* un_bf = (__bf16*)ws;                                // 64 MB
  float*  uB    = (float*)(ws + ROWS * DIN * 2);              // 64 MB
  __bf16* Bt    = (__bf16*)(ws + ROWS * DIN * 2 + ROWS * STATE * 4);  // [512][1024]
  __bf16* Ct    = Bt + (long)DIN * STATE;                     // [1024][512]
  __bf16* xbf   = un_bf;  // reuse un buffer after GEMM1 (32 MB needed)

  // 1) weights -> bf16, transposed so GEMM operands are contiguous along K
  cvt_tr_kernel<<<(DIN * STATE + 255) / 256, 256, 0, stream>>>(B, Bt, DIN, STATE);
  cvt_tr_kernel<<<(STATE * DOUT + 255) / 256, 256, 0, stream>>>(C, Ct, STATE, DOUT);

  // 2) rmsnorm -> bf16
  rmsnorm_kernel<<<(int)ROWS, 256, 0, stream>>>(u, nw, un_bf, DIN);

  // 3) uB = un @ B        [32768,1024] @ [1024,512]
  {
    dim3 grid((int)(ROWS / 128), STATE / 128);
    gemm_bf16_kernel<<<grid, 256, 0, stream>>>(un_bf, Bt, uB,
                                               (int)ROWS, STATE, DIN);
  }

  // 4) rotation scan (writes xbf + new_state)
  scan_kernel<<<BATCH, K2, 0, stream>>>(uB, A, x0, xbf, new_state, L, K2);

  // 5) y = x @ C          [32768,512] @ [512,1024]
  {
    dim3 grid((int)(ROWS / 128), DOUT / 128);
    gemm_bf16_kernel<<<grid, 256, 0, stream>>>(xbf, Ct, y,
                                               (int)ROWS, DOUT, STATE);
  }
}
